// GCN_simple_71743133712656
// MI455X (gfx1250) — matrix operand order repaired and checked by reading, not compiled, not run on hardware
//
#include <hip/hip_runtime.h>

// Problem constants (from reference): B=32, N=2000, F=H1=H2=32, L=10.
#define B_   32
#define N_   2000
#define F_   32
#define L_   10
#define KT_  63     // ceil(2000 / 32) K-tiles
#define RT_  125    // 2000 / 16 row tiles per batch

typedef __attribute__((ext_vector_type(16))) __bf16 v16bf;
typedef __attribute__((ext_vector_type(8)))  float  v8f;

union BFRegs { unsigned int u[8]; v16bf b; };

// f32x2 -> packed bf16x2, round-to-nearest (ties up), 3 VALU ops:
// two v_add_u32 + one v_perm_b32 selecting the high halves.
__device__ __forceinline__ unsigned int pack2(float lo, float hi) {
  unsigned int ul = __float_as_uint(lo) + 0x8000u;
  unsigned int uh = __float_as_uint(hi) + 0x8000u;
  return __builtin_amdgcn_perm(uh, ul, 0x07060302u);
}

// ---------------------------------------------------------------------------
// Kernel 1/3: XW = src @ W  (src: (B,N,32) f32, W: (32,32) f32), written as the
// bf16 WMMA B-operand register image:
//   P[b][t][ftile][lane][j]  (uint32 = two packed bf16)
//   lane<16 : N = ftile*16 + lane,       K = 32t + 16*(j>>2) + 2*(j&3), +1
//   lane>=16: N = ftile*16 + (lane-16),  K = 32t + 8 + 16*(j>>2) + 2*(j&3), +1
// K >= 2000 entries are zero (tail tile).
// ---------------------------------------------------------------------------
__global__ void __launch_bounds__(256)
gcn_pack_xw(const float* __restrict__ src, const float* __restrict__ W,
            unsigned int* __restrict__ P) {
  const int t = blockIdx.x;    // K-tile
  const int b = blockIdx.y;    // batch
  const int tid = threadIdx.x;

  __shared__ float xs[32][33];   // 32 source rows (k) x 32 features
  __shared__ float ws[32][33];   // W[f][n]
  for (int i = tid; i < 1024; i += 256) {
    int r = i >> 5, c = i & 31;
    int k = t * 32 + r;
    xs[r][c] = (k < N_) ? src[((size_t)b * N_ + k) * F_ + c] : 0.0f;
    ws[r][c] = W[i];
  }
  __syncthreads();

  const int lane  = tid >> 3;
  const int j     = tid & 7;
  const int khalf = lane >> 4;
  const int r0    = khalf * 8 + (j & 3) * 2 + (j >> 2) * 16;   // local k row (even)

  for (int ft = 0; ft < 2; ++ft) {
    const int n = ft * 16 + (lane & 15);
    float d0 = 0.0f, d1 = 0.0f;
#pragma unroll
    for (int f = 0; f < 32; ++f) {
      float w = ws[f][n];
      d0 += xs[r0][f] * w;
      d1 += xs[r0 + 1][f] * w;
    }
    P[((((size_t)b * KT_ + t) * 2 + ft) * 32 + lane) * 8 + j] = pack2(d0, d1);
  }
}

// ---------------------------------------------------------------------------
// Kernel 2/4: Y = relu(adj[b] @ X[b]) with bf16 WMMA, f32 accumulate.
// One wave per 16-row output tile; two accumulators (features 0-15, 16-31).
// A-operand lane layout (16-bit A 16x32):
//   lane<16 : M = lane,    K = {0..7} in u[0..3], {16..23} in u[4..7]
//   lane>=16: M = lane-16, K = {8..15} in u[0..3], {24..31} in u[4..7]
// ---------------------------------------------------------------------------
__global__ void __launch_bounds__(256)
gcn_spmm_relu(const float* __restrict__ adj, const unsigned int* __restrict__ P,
              float* __restrict__ out) {
  const int lane    = threadIdx.x & 31;
  const int w       = blockIdx.x * 8 + (threadIdx.x >> 5);
  const int b       = w / RT_;
  const int rowbase = (w % RT_) * 16;
  const int halfsel = lane >> 4;
  const int mrow    = lane & 15;

  const float* __restrict__ arow = adj + ((size_t)b * N_ + rowbase + mrow) * N_;
  const uint4* __restrict__ pB   = (const uint4*)P;

  v8f c0 = {};
  v8f c1 = {};

  // Main loop: 62 full K-tiles, branch-free body.
#pragma unroll 2
  for (int t = 0; t < KT_ - 1; ++t) {
    const int kb = t * 32 + halfsel * 8;

    float4 a0 = *(const float4*)(arow + kb);
    float4 a1 = *(const float4*)(arow + kb + 4);
    float4 a2 = *(const float4*)(arow + kb + 16);
    float4 a3 = *(const float4*)(arow + kb + 20);
    __builtin_prefetch(arow + kb + 32, 0, 3);   // next K-tile (global_prefetch_b8)

    BFRegs A;
    A.u[0] = pack2(a0.x, a0.y); A.u[1] = pack2(a0.z, a0.w);
    A.u[2] = pack2(a1.x, a1.y); A.u[3] = pack2(a1.z, a1.w);
    A.u[4] = pack2(a2.x, a2.y); A.u[5] = pack2(a2.z, a2.w);
    A.u[6] = pack2(a3.x, a3.y); A.u[7] = pack2(a3.z, a3.w);

    // Pre-packed B operands: 32 contiguous bytes per lane, coalesced.
    const size_t base = (((size_t)b * KT_ + t) * 2 * 32 + lane) * 2;  // uint4 index
    BFRegs B0, B1;
    uint4 q0 = pB[base],      q1 = pB[base + 1];       // ftile 0
    uint4 q2 = pB[base + 64], q3 = pB[base + 65];      // ftile 1
    B0.u[0] = q0.x; B0.u[1] = q0.y; B0.u[2] = q0.z; B0.u[3] = q0.w;
    B0.u[4] = q1.x; B0.u[5] = q1.y; B0.u[6] = q1.z; B0.u[7] = q1.w;
    B1.u[0] = q2.x; B1.u[1] = q2.y; B1.u[2] = q2.z; B1.u[3] = q2.w;
    B1.u[4] = q3.x; B1.u[5] = q3.y; B1.u[6] = q3.z; B1.u[7] = q3.w;

    c0 = __builtin_amdgcn_wmma_f32_16x16x32_bf16(false, A.b, false, B0.b,
                                                 (short)0, c0, false, false);
    c1 = __builtin_amdgcn_wmma_f32_16x16x32_bf16(false, A.b, false, B1.b,
                                                 (short)0, c1, false, false);
  }

  // Peeled K tail (t = 62): k in [1984, 2000) valid, upper half zero.
  {
    const int t = KT_ - 1;
    const int kb = t * 32 + halfsel * 8;

    float4 a0 = *(const float4*)(arow + kb);
    float4 a1 = *(const float4*)(arow + kb + 4);

    BFRegs A;
    A.u[0] = pack2(a0.x, a0.y); A.u[1] = pack2(a0.z, a0.w);
    A.u[2] = pack2(a1.x, a1.y); A.u[3] = pack2(a1.z, a1.w);
    A.u[4] = 0u; A.u[5] = 0u; A.u[6] = 0u; A.u[7] = 0u;

    const size_t base = (((size_t)b * KT_ + t) * 2 * 32 + lane) * 2;
    BFRegs B0, B1;
    uint4 q0 = pB[base],      q1 = pB[base + 1];
    uint4 q2 = pB[base + 64], q3 = pB[base + 65];
    B0.u[0] = q0.x; B0.u[1] = q0.y; B0.u[2] = q0.z; B0.u[3] = q0.w;
    B0.u[4] = q1.x; B0.u[5] = q1.y; B0.u[6] = q1.z; B0.u[7] = q1.w;
    B1.u[0] = q2.x; B1.u[1] = q2.y; B1.u[2] = q2.z; B1.u[3] = q2.w;
    B1.u[4] = q3.x; B1.u[5] = q3.y; B1.u[6] = q3.z; B1.u[7] = q3.w;

    c0 = __builtin_amdgcn_wmma_f32_16x16x32_bf16(false, A.b, false, B0.b,
                                                 (short)0, c0, false, false);
    c1 = __builtin_amdgcn_wmma_f32_16x16x32_bf16(false, A.b, false, B1.b,
                                                 (short)0, c1, false, false);
  }

  // C/D layout: VGPR j, lanes 0-15 -> M=j, N=lane; lanes 16-31 -> M=8+j, N=lane-16.
  const int rw = rowbase + halfsel * 8;
  float* __restrict__ orow = out + ((size_t)b * N_ + rw) * F_ + mrow;
#pragma unroll
  for (int j = 0; j < 8; ++j) {
    orow[(size_t)j * F_]      = fmaxf(c0[j], 0.0f);
    orow[(size_t)j * F_ + 16] = fmaxf(c1[j], 0.0f);
  }
}

// ---------------------------------------------------------------------------
// Kernel 5: out[b,l] = out_b[l] + sum_n (sum_f X[b,n,f]) * out_w[l,n]
// ---------------------------------------------------------------------------
__global__ void __launch_bounds__(256)
gcn_pool_out(const float* __restrict__ X, const float* __restrict__ out_w,
             const float* __restrict__ out_b, float* __restrict__ out) {
  const int b = blockIdx.x;
  const int tid = threadIdx.x;

  float acc[L_];
#pragma unroll
  for (int l = 0; l < L_; ++l) acc[l] = 0.0f;

  for (int n = tid; n < N_; n += 256) {
    const float* row = X + ((size_t)b * N_ + n) * F_;
    float s = 0.0f;
#pragma unroll
    for (int f = 0; f < F_; ++f) s += row[f];
#pragma unroll
    for (int l = 0; l < L_; ++l) acc[l] += s * out_w[(size_t)l * N_ + n];
  }

  __shared__ float red[L_][256];
#pragma unroll
  for (int l = 0; l < L_; ++l) red[l][tid] = acc[l];
  for (int off = 128; off > 0; off >>= 1) {
    __syncthreads();
    if (tid < off)
#pragma unroll
      for (int l = 0; l < L_; ++l) red[l][tid] += red[l][tid + off];
  }
  __syncthreads();
  if (tid < L_) out[b * L_ + tid] = red[tid][0] + out_b[tid];
}

// ---------------------------------------------------------------------------
extern "C" void kernel_launch(void* const* d_in, const int* in_sizes, int n_in,
                              void* d_out, int out_size, void* d_ws, size_t ws_size,
                              hipStream_t stream) {
  const float* v     = (const float*)d_in[0];   // (32, 2000, 32)
  const float* adj   = (const float*)d_in[1];   // (32, 2000, 2000)
  const float* W1    = (const float*)d_in[2];   // (32, 32)
  const float* W2    = (const float*)d_in[3];   // (32, 32)
  const float* out_w = (const float*)d_in[4];   // (10, 2000)
  const float* out_b = (const float*)d_in[5];   // (10,)
  float* out = (float*)d_out;                   // (32, 10)

  // Workspace layout: packed bf16 B-operands, then f32 hidden activations.
  const size_t P_BYTES = (size_t)B_ * KT_ * 2 * 32 * 8 * 4;      // ~4.13 MB
  const size_t X_OFF   = (P_BYTES + 255) & ~(size_t)255;
  unsigned int* P = (unsigned int*)d_ws;
  float*        X = (float*)((char*)d_ws + X_OFF);               // ~8.19 MB

  dim3 gPack(KT_, B_);
  const int layerBlocks = (B_ * RT_) / 8;   // 4000 waves / 8 waves per block

  gcn_pack_xw  <<<gPack,       256, 0, stream>>>(v, W1, P);       // XW1
  gcn_spmm_relu<<<layerBlocks, 256, 0, stream>>>(adj, P, X);      // X1 = relu(adj@XW1)
  gcn_pack_xw  <<<gPack,       256, 0, stream>>>(X, W2, P);       // XW2
  gcn_spmm_relu<<<layerBlocks, 256, 0, stream>>>(adj, P, X);      // X2 = relu(adj@XW2)
  gcn_pool_out <<<B_,          256, 0, stream>>>(X, out_w, out_b, out);
}
